// ComplexConv2Deffangle_30906584662520
// MI455X (gfx1250) — matrix operand order repaired and checked by reading, not compiled, not run on hardware
//
#include <hip/hip_runtime.h>
#include <hip/hip_bf16.h>

typedef _Float16 h16 __attribute__((ext_vector_type(16)));
typedef float    f8  __attribute__((ext_vector_type(8)));

#define EPS   1e-6f
#define NB    32
#define OXY   62
#define NPIX  3844      // 62*62
#define C_OUT 128

// workspace layout (float indices unless noted)
#define W1G_OFF 0                 // 576 floats
#define W1R_OFF 576               // 576 floats
#define WA_OFF  1152              // f16 area: 16384 halves = 32768 B (byte off 4608, 32B aligned)
#define WA_HALVES 16384
#define LX_OFF  9344              // = (4608+32768)/4 ; 8388608 floats of log(x_abs+eps)
#define LX_ELEMS (NB*64*64*64)    // 8388608
#define WS_NEED_FULL ((size_t)LX_OFF*4 + (size_t)LX_ELEMS*4)   // 33,591,808 B

// ---------------- weight preparation ----------------
__global__ __launch_bounds__(256) void k_prep(const float* __restrict__ w1,
                                              const float* __restrict__ w2,
                                              float* __restrict__ ws_f) {
  __shared__ float rs1[64];
  __shared__ float rs2[128];
  __shared__ float tot[2];
  float* w1g = ws_f + W1G_OFF;
  float* w1r = ws_f + W1R_OFF;
  _Float16* wA = (_Float16*)(ws_f + WA_OFF);
  int t = threadIdx.x;
  if (t < 64) {
    float s = 0.f;
    for (int k = 0; k < 9; ++k) { float v = w1[t*9+k]; s += v*v; }
    rs1[t] = s;
  }
  if (t < 128) {
    float s = 0.f;
    for (int c = 0; c < 64; ++c) { float v = w2[t*64+c]; s += v*v; }
    rs2[t] = s;
  }
  __syncthreads();
  if (t == 0)  { float s = 0.f; for (int i = 0; i < 64;  ++i) s += rs1[i]; tot[0] = s; }
  if (t == 32) { float s = 0.f; for (int i = 0; i < 128; ++i) s += rs2[i]; tot[1] = s; }
  __syncthreads();
  if (t < 64) {
    for (int k = 0; k < 9; ++k) {
      float v = w1[t*9+k]; v *= v;
      w1g[t*9+k] = v / tot[0];
      w1r[t*9+k] = v / rs1[t];
    }
  }
  // Pack w2 (glob- and row-normalized) as f16 WMMA A-fragments.
  // 16-bit A 16x32 layout: lane = m + 16*g ; element s holds
  // k = g*16 + s (contiguous, matching the B fragment reads).
  for (int j = t; j < WA_HALVES; j += 256) {
    int frag = j >> 9, within = j & 511;
    int lane = within >> 4, hv = within & 15;
    int v = hv >> 1, lo = hv & 1, g = lane >> 4, m = lane & 15;
    int mtile = frag >> 2, branch = (frag >> 1) & 1, kbi = frag & 1;
    int ch = mtile * 16 + m;
    int k = kbi * 32 + (g << 4) + hv;
    float w = w2[ch * 64 + k]; w *= w;
    wA[j] = (_Float16)(w / (branch ? rs2[ch] : tot[1]));
  }
}

// ---------------- log precompute (vectorized b128) ----------------
__global__ __launch_bounds__(256) void k_log(const float* __restrict__ x,
                                             float* __restrict__ lx) {
  size_t j = ((size_t)blockIdx.x * 256 + threadIdx.x) * 4;   // 4 elems/thread
  size_t b = j >> 18;
  size_t rest = j & ((1 << 18) - 1);
  const float4 v = *(const float4*)(x + (b << 19) + (1 << 18) + rest);
  float4 r;
  r.x = logf(v.x + EPS); r.y = logf(v.y + EPS);
  r.z = logf(v.z + EPS); r.w = logf(v.w + EPS);
  *(float4*)(lx + j) = r;
}

// ---------------- fused depthwise + WMMA pointwise ----------------
template <bool USE_LX>
__global__ __launch_bounds__(256) void k_main(const float* __restrict__ x,
                                              const float* __restrict__ ws_f,
                                              float* __restrict__ out) {
  // [branch][pixel][channel]; 72-half stride -> rows 144 B (16B-aligned -> ds_load_b128)
  __shared__ __align__(16) _Float16 ldsB[2][64][72];
  const float* w1g = ws_f + W1G_OFF;
  const float* w1r = ws_f + W1R_OFF;
  const float* lx  = ws_f + LX_OFF;
  const _Float16* wA = (const _Float16*)(ws_f + WA_OFF);

  int b   = blockIdx.x / OXY;
  int row = blockIdx.x % OXY;
  int t = threadIdx.x;

  // Phase 1: fp32 depthwise 3x3, two adjacent channels per thread, packed b32 LDS stores
  #pragma unroll
  for (int i = 0; i < 16; ++i) {
    int p = i * 256 + t;                 // 0..4095 channel-pairs
    int px = p & 63;
    int cpair = (p >> 6) & 31;
    int branch = p >> 11;                // uniform per wave per iteration
    int c0 = cpair << 1;
    float a0 = 0.f, a1 = 0.f;
    if (px < OXY) {
      const float* wr = (branch ? w1r : w1g) + c0 * 9;
      const float* src;
      if (branch == 0)      src = x  + ((size_t)b << 19);
      else if (USE_LX)      src = lx + ((size_t)b << 18);
      else                  src = x  + ((size_t)b << 19) + (1 << 18);
      const float* p0 = src + ((size_t)c0 << 12) + row * 64 + px;
      #pragma unroll
      for (int dr = 0; dr < 3; ++dr)
        #pragma unroll
        for (int dc = 0; dc < 3; ++dc) {
          float v0 = p0[dr * 64 + dc];
          float v1 = p0[4096 + dr * 64 + dc];
          if (branch && !USE_LX) { v0 = logf(v0 + EPS); v1 = logf(v1 + EPS); }
          a0 = fmaf(wr[dr * 3 + dc],     v0, a0);
          a1 = fmaf(wr[9 + dr * 3 + dc], v1, a1);
        }
    }
    union { _Float16 h[2]; unsigned int u; } pk;
    pk.h[0] = (_Float16)a0; pk.h[1] = (_Float16)a1;
    ((unsigned int*)ldsB)[((branch << 6) + px) * 36 + cpair] = pk.u;
  }
  __syncthreads();

  // Phase 2: per-wave 16x64 * 64x64 GEMM via v_wmma_f32_16x16x32_f16
  int wave = t >> 5, lane = t & 31;
  int g = lane >> 4, n = lane & 15;
  #pragma unroll
  for (int branch = 0; branch < 2; ++branch) {
    h16 a0 = *(const h16*)(wA + ((size_t)((wave * 4 + branch * 2 + 0) * 32 + lane)) * 16);
    h16 a1 = *(const h16*)(wA + ((size_t)((wave * 4 + branch * 2 + 1) * 32 + lane)) * 16);
    #pragma unroll
    for (int nt = 0; nt < 4; ++nt) {
      int px = nt * 16 + n;
      // B 32x16 k order used here (matches A packing): lanes 0-15 hold K=0..15, lanes 16-31 hold K=16..31.
      // Per lane the 8 dwords per K-block are contiguous -> two ds_load_b128 each.
      const unsigned int* brow = (const unsigned int*)&ldsB[branch][px][0];
      const uint4* bq = (const uint4*)(brow + (g << 3));
      union { h16 v; uint4 q[2]; } b0, b1;
      b0.q[0] = bq[0]; b0.q[1] = bq[1];   // K 0..31 halves for this lane-group
      b1.q[0] = bq[4]; b1.q[1] = bq[5];   // K 32..63
      f8 acc = {};
      acc = __builtin_amdgcn_wmma_f32_16x16x32_f16(false, a0, false, b0.v, (short)0, acc, false, false);
      acc = __builtin_amdgcn_wmma_f32_16x16x32_f16(false, a1, false, b1.v, (short)0, acc, false, false);

      size_t base = ((((size_t)b * 2 + branch) * C_OUT + wave * 16) * NPIX) + (size_t)row * OXY;
      #pragma unroll
      for (int r = 0; r < 8; ++r) {
        int m = r + (g << 3);             // C/D layout: vgpr r, lanes 16-31 -> M = r+8
        float val = acc[r];
        if (branch) val = expf(val);
        if (px < OXY) out[base + (size_t)m * NPIX + px] = val;
      }
    }
  }
}

extern "C" void kernel_launch(void* const* d_in, const int* in_sizes, int n_in,
                              void* d_out, int out_size, void* d_ws, size_t ws_size,
                              hipStream_t stream) {
  (void)in_sizes; (void)n_in; (void)out_size;
  const float* x  = (const float*)d_in[0];
  const float* w1 = (const float*)d_in[1];
  const float* w2 = (const float*)d_in[2];
  float* out  = (float*)d_out;
  float* ws_f = (float*)d_ws;

  k_prep<<<1, 256, 0, stream>>>(w1, w2, ws_f);

  if (ws_size >= WS_NEED_FULL) {
    k_log<<<LX_ELEMS / (256 * 4), 256, 0, stream>>>(x, ws_f + LX_OFF);
    k_main<true><<<NB * OXY, 256, 0, stream>>>(x, ws_f, out);
  } else {
    k_main<false><<<NB * OXY, 256, 0, stream>>>(x, ws_f, out);
  }
}
